// SelfAttention_59717225284222
// MI455X (gfx1250) — compile-verified, matrix operand order checked by reading, not run on hardware
//
#include <hip/hip_runtime.h>
#include <hip/hip_bf16.h>

// ---------------------------------------------------------------------------
// Problem constants (match reference)
// ---------------------------------------------------------------------------
#define BB 4
#define CC 256
#define LL 8192
#define HH 4
#define DD 64
#define MTOT (BB * LL)   // 32768 tokens

typedef __attribute__((ext_vector_type(16))) __bf16 v16bf;
typedef __attribute__((ext_vector_type(8)))  float  v8f;

union FragU { v16bf v; uint4 u[2]; __bf16 h[16]; };
union Pack8 { uint4 u; __bf16 h[8]; };

// Standard row-major fragment: element (r, k) = base[r*stride + k-pattern]
// Matches CDNA5 16-bit 16x32 A/B VGPR layout: lane = r + 16*((k>>3)&1),
// vector elem = (k&7) + 8*((k>>4)&1).  Two 16-byte ds_load_b128 per lane.
__device__ inline v16bf frag_ld(const __bf16* base, int stride) {
  const int t = threadIdx.x & 31;
  const int r = t & 15;
  const int hi = t >> 4;
  FragU f;
  const __bf16* p = base + r * stride + hi * 8;
  f.u[0] = *(const uint4*)(p);
  f.u[1] = *(const uint4*)(p + 16);
  return f.v;
}

#define WMMA_BF16(A, Bf, Cacc) \
  __builtin_amdgcn_wmma_f32_16x16x32_bf16(false, (A), false, (Bf), (short)0, (Cacc), false, false)

// ---------------------------------------------------------------------------
// Async global -> LDS copies (CDNA5 GLOBAL_LOAD_ASYNC_TO_LDS_B128, ASYNCcnt).
// Generic pointers to __shared__ carry the LDS byte offset in their low 32
// bits (ISA 10.2 aperture mapping), which is exactly the VDST operand.
// ---------------------------------------------------------------------------
__device__ inline void async_copy_b128(void* lds_ptr, const void* gptr) {
  unsigned loff = (unsigned)(unsigned long long)lds_ptr;
  asm volatile("global_load_async_to_lds_b128 %0, %1, off"
               :: "v"(loff), "v"(gptr) : "memory");
}

__device__ inline void async_wait0() {
#if __has_builtin(__builtin_amdgcn_s_wait_asynccnt)
  __builtin_amdgcn_s_wait_asynccnt(0);
#else
  asm volatile("s_wait_asynccnt 0x0" ::: "memory");
#endif
}

// ---------------------------------------------------------------------------
// Small helper kernels
// ---------------------------------------------------------------------------
__global__ void cvt_f32_bf16_kernel(const float* __restrict__ src,
                                    __bf16* __restrict__ dst, int n) {
  int i = blockIdx.x * blockDim.x + threadIdx.x;
  if (i < n) dst[i] = (__bf16)src[i];
}

__global__ void zero_f32_kernel(float* __restrict__ p, int n) {
  int i = blockIdx.x * blockDim.x + threadIdx.x;
  if (i < n) p[i] = 0.0f;
}

// x [B,C,L] f32  ->  cat[:, 0:256] bf16 (row = token, ld = 512)
__global__ __launch_bounds__(256)
void transpose_kernel(const float* __restrict__ x, __bf16* __restrict__ cat) {
  __shared__ float t[32][33];
  const int tx = threadIdx.x, ty = threadIdx.y;
  const int l0 = blockIdx.x * 32, c0 = blockIdx.y * 32, b = blockIdx.z;
#pragma unroll
  for (int i = 0; i < 4; ++i) {
    int c = ty + i * 8;
    t[c][tx] = x[((size_t)b * CC + c0 + c) * LL + l0 + tx];
  }
  __syncthreads();
#pragma unroll
  for (int i = 0; i < 4; ++i) {
    int lr = ty + i * 8;
    cat[(size_t)(b * LL + l0 + lr) * 512 + c0 + tx] = (__bf16)t[tx][lr];
  }
}

// ---------------------------------------------------------------------------
// Generic 128x256 bf16 WMMA GEMM:  Out[m, n0+n] = epi( sum_k A[m,k]*W[n0+n,k] )
// Modes: Q/K -> elu(x+bias)+1 ; V -> (x+bias)/L ; RELU -> max(x,0)
// Mode K also accumulates column sums (K.sum over tokens) into KsumG[b*256+n].
// ---------------------------------------------------------------------------
enum { MODE_Q = 0, MODE_K = 1, MODE_V = 2, MODE_RELU = 3 };

template <int MODE>
__global__ __launch_bounds__(256)
void gemm_bf16_kernel(const __bf16* __restrict__ A, int lda, int Ktot,
                      const __bf16* __restrict__ Bw,
                      const float* __restrict__ bias,
                      __bf16* __restrict__ Out, int ldo,
                      float* __restrict__ KsumG) {
  __shared__ __bf16 Alds[128 * 40];
  __shared__ __bf16 Blds[256 * 40];
  __shared__ float  ksumLds[256];

  const int tid  = threadIdx.x;
  const int wave = tid >> 5;
  const int lane = tid & 31;
  const int r16  = lane & 15;
  const int hi   = lane >> 4;
  const int token0 = blockIdx.x * 128;
  const int n0     = blockIdx.y * 256;
  const int mb     = wave * 16;

  if constexpr (MODE == MODE_K) ksumLds[tid] = 0.0f;

  const v8f vz = {0.f, 0.f, 0.f, 0.f, 0.f, 0.f, 0.f, 0.f};
  v8f acc[16];
#pragma unroll
  for (int j = 0; j < 16; ++j) acc[j] = vz;

  for (int k0 = 0; k0 < Ktot; k0 += 32) {
    __syncthreads();
#pragma unroll
    for (int it = 0; it < 2; ++it) {   // A tile 128x32 (async -> LDS)
      int idx = tid * 2 + it;
      int r = idx >> 2, part = idx & 3;
      async_copy_b128(&Alds[r * 40 + part * 8],
                      A + (size_t)(token0 + r) * lda + k0 + part * 8);
    }
#pragma unroll
    for (int it = 0; it < 4; ++it) {   // B tile 256x32 (async -> LDS)
      int idx = tid * 4 + it;
      int r = idx >> 2, part = idx & 3;
      async_copy_b128(&Blds[r * 40 + part * 8],
                      Bw + (size_t)(n0 + r) * Ktot + k0 + part * 8);
    }
    if (k0 + 32 < Ktot)  // global_prefetch_b8 of next A slice
      __builtin_prefetch(A + (size_t)(token0 + (tid >> 1)) * lda + k0 + 32, 0, 3);
    async_wait0();
    __syncthreads();
    v16bf fa = frag_ld(&Alds[mb * 40], 40);
#pragma unroll
    for (int j = 0; j < 16; ++j) {
      v16bf fb = frag_ld(&Blds[j * 16 * 40], 40);
      acc[j] = WMMA_BF16(fa, fb, acc[j]);
    }
  }

  // epilogue
#pragma unroll
  for (int j = 0; j < 16; ++j) {
    const int n = j * 16 + r16;
    float colsum = 0.0f;
#pragma unroll
    for (int i = 0; i < 8; ++i) {
      const int m = mb + hi * 8 + i;
      float v = acc[j][i];
      if constexpr (MODE == MODE_RELU) {
        v = fmaxf(v, 0.0f);
      } else {
        v += bias[n];
        if constexpr (MODE == MODE_Q || MODE == MODE_K)
          v = (v > 0.0f) ? (v + 1.0f) : __expf(v);   // elu(x)+1
        else
          v *= (1.0f / (float)LL);                    // V / v_len
      }
      Out[(size_t)(token0 + m) * ldo + n0 + n] = (__bf16)v;
      if constexpr (MODE == MODE_K) colsum += v;
    }
    if constexpr (MODE == MODE_K) {
      colsum += __shfl_xor(colsum, 16);      // fold both 8-row halves
      if (hi == 0) atomicAdd(&ksumLds[n], colsum);   // ds_add_f32
    }
  }
  if constexpr (MODE == MODE_K) {
    __syncthreads();
    const int b = token0 >> 13;   // token0 / 8192
    atomicAdd(&KsumG[b * 256 + tid], ksumLds[tid]);
  }
}

// ---------------------------------------------------------------------------
// KV[b,h,d,e] = sum_l K[l,d] * V[l,e]   (per (b,h): 64x64, K-dim = L)
// K/V tiles are written into LDS transposed ([d][l] / [e][l]) so fragment
// loads use the vectorized ds_load_b128 path.
// grid.x = B*H, grid.y = L/1024 chunks; fp32 global atomics combine chunks.
// ---------------------------------------------------------------------------
__global__ __launch_bounds__(256)
void kv_kernel(const __bf16* __restrict__ Kbuf, const __bf16* __restrict__ Vbuf,
               float* __restrict__ KV) {
  __shared__ __bf16 Ktlds[64 * 40];   // [d][l]
  __shared__ __bf16 Vtlds[64 * 40];   // [e][l]
  const int tid  = threadIdx.x;
  const int wave = tid >> 5;
  const int lane = tid & 31;
  const int r16  = lane & 15;
  const int hi   = lane >> 4;
  const int bh = blockIdx.x;              // 0..15
  const int b  = bh >> 2, h = bh & 3;
  const int l0 = blockIdx.y * 1024;
  const int tile0 = wave * 2;             // 8 waves x 2 of 16 (d,e) tiles

  const v8f vz = {0.f, 0.f, 0.f, 0.f, 0.f, 0.f, 0.f, 0.f};
  v8f acc[2];
  acc[0] = vz; acc[1] = vz;

  for (int l = l0; l < l0 + 1024; l += 32) {
    __syncthreads();
    {
      const int r = tid >> 3, part = tid & 7;   // row l+r, 8 d's per thread
      Pack8 kq, vq;
      kq.u = *(const uint4*)(Kbuf + (size_t)(b * LL + l + r) * CC + h * DD + part * 8);
      vq.u = *(const uint4*)(Vbuf + (size_t)(b * LL + l + r) * CC + h * DD + part * 8);
#pragma unroll
      for (int j = 0; j < 8; ++j) {             // transposed LDS scatter
        Ktlds[(part * 8 + j) * 40 + r] = kq.h[j];
        Vtlds[(part * 8 + j) * 40 + r] = vq.h[j];
      }
    }
    __syncthreads();
#pragma unroll
    for (int t = 0; t < 2; ++t) {
      const int id = tile0 + t, dt = id >> 2, et = id & 3;
      v16bf fa = frag_ld(&Ktlds[dt * 16 * 40], 40);   // A[d, l] = K[l, d]
      v16bf fb = frag_ld(&Vtlds[et * 16 * 40], 40);   // Bt[e, l] = V[l, e]
      acc[t] = WMMA_BF16(fa, fb, acc[t]);
    }
  }
#pragma unroll
  for (int t = 0; t < 2; ++t) {
    const int id = tile0 + t, dt = id >> 2, et = id & 3;
#pragma unroll
    for (int i = 0; i < 8; ++i) {
      const int d = dt * 16 + hi * 8 + i;
      const int e = et * 16 + r16;
      atomicAdd(&KV[((size_t)bh * DD + d) * DD + e], acc[t][i]);
    }
  }
}

// ---------------------------------------------------------------------------
// Z[token,h] = L / (sparsify(Q[token,h,:] . Ksum[b,h,:]) + eps)
// ---------------------------------------------------------------------------
__global__ __launch_bounds__(256)
void zscore_kernel(const __bf16* __restrict__ Qbuf, const float* __restrict__ Ksum,
                   const float* __restrict__ thr_p, float* __restrict__ Zbuf) {
  const int g = blockIdx.x * blockDim.x + threadIdx.x;   // 0..131071
  const int token = g >> 2, h = g & 3;
  const int b = token >> 13;
  const float thr = thr_p[0];
  const __bf16* q = Qbuf + (size_t)token * CC + h * DD;
  const float* ks = Ksum + b * 256 + h * DD;
  float s = 0.0f;
#pragma unroll
  for (int c = 0; c < 8; ++c) {          // 8 x uint4 = 64 bf16
    Pack8 qp;
    qp.u = *(const uint4*)(q + c * 8);
#pragma unroll
    for (int j = 0; j < 8; ++j) s += (float)qp.h[j] * ks[c * 8 + j];
  }
  const float sp = (s > thr) ? s : 0.0f;
  Zbuf[g] = (float)LL / (sp + 1e-6f);
}

// ---------------------------------------------------------------------------
// out = LN1( (Q @ KV_h) * Z ) -> cat[:, 256:512] bf16
// ---------------------------------------------------------------------------
__global__ __launch_bounds__(256)
void attn_kernel(const __bf16* __restrict__ Qbuf, const float* __restrict__ KV,
                 const float* __restrict__ Zbuf,
                 const float* __restrict__ g1, const float* __restrict__ b1,
                 __bf16* __restrict__ cat) {
  __shared__ __bf16 Alds[128 * 40];
  __shared__ __bf16 kvt[HH * DD * DD];   // kvt[h][e][d] = KV[b,h,d,e] (transposed)
  __shared__ float  Zl[128 * 4];
  __shared__ float  gl[256], bl[256];

  const int tid  = threadIdx.x;
  const int wave = tid >> 5;
  const int lane = tid & 31;
  const int r16  = lane & 15;
  const int hi   = lane >> 4;
  const int token0 = blockIdx.x * 128;
  const int b      = token0 >> 13;
  const int mb     = wave * 16;

  // KVT fill: float4 global reads, transposed bf16 LDS scatter
  for (int it = 0; it < 16; ++it) {
    const int idx = it * 256 + tid;        // 4096 groups of 4 e's
    const int h = idx >> 10, rem = idx & 1023, d = rem >> 4, e4 = rem & 15;
    const float4 kvv =
        *(const float4*)(KV + (((size_t)(b * HH + h) * DD + d) * DD) + e4 * 4);
    kvt[h * 4096 + (e4 * 4 + 0) * DD + d] = (__bf16)kvv.x;
    kvt[h * 4096 + (e4 * 4 + 1) * DD + d] = (__bf16)kvv.y;
    kvt[h * 4096 + (e4 * 4 + 2) * DD + d] = (__bf16)kvv.z;
    kvt[h * 4096 + (e4 * 4 + 3) * DD + d] = (__bf16)kvv.w;
  }
#pragma unroll
  for (int it = 0; it < 2; ++it) {
    const int z = it * 256 + tid;
    Zl[z] = Zbuf[(size_t)token0 * 4 + z];
  }
  gl[tid] = g1[tid];
  bl[tid] = b1[tid];

  const v8f vz = {0.f, 0.f, 0.f, 0.f, 0.f, 0.f, 0.f, 0.f};
  v8f acc[16];
#pragma unroll
  for (int j = 0; j < 16; ++j) acc[j] = vz;

  for (int h = 0; h < 4; ++h) {
    for (int s = 0; s < 2; ++s) {
      __syncthreads();
      const int k0 = h * DD + s * 32;
#pragma unroll
      for (int it = 0; it < 2; ++it) {        // Q tile 128x32 (async -> LDS)
        int idx = tid * 2 + it;
        int r = idx >> 2, part = idx & 3;
        async_copy_b128(&Alds[r * 40 + part * 8],
                        Qbuf + (size_t)(token0 + r) * CC + k0 + part * 8);
      }
      async_wait0();
      __syncthreads();
      v16bf fa = frag_ld(&Alds[mb * 40], 40);
#pragma unroll
      for (int jj = 0; jj < 4; ++jj) {
        const int j = h * 4 + jj;
        v16bf fb = frag_ld(&kvt[h * 4096 + (jj * 16) * DD + s * 32], DD);
        acc[j] = WMMA_BF16(fa, fb, acc[j]);
      }
    }
  }

  // scale by Z, then LayerNorm over C=256 via cross-lane shuffles
  float s1[8] = {0, 0, 0, 0, 0, 0, 0, 0};
  float s2[8] = {0, 0, 0, 0, 0, 0, 0, 0};
#pragma unroll
  for (int j = 0; j < 16; ++j) {
    const int h = j >> 2;
#pragma unroll
    for (int i = 0; i < 8; ++i) {
      const int m = mb + hi * 8 + i;
      float v = acc[j][i] * Zl[m * 4 + h];
      acc[j][i] = v;
      s1[i] += v;
      s2[i] += v * v;
    }
  }
#pragma unroll
  for (int i = 0; i < 8; ++i) {
#pragma unroll
    for (int off = 1; off < 16; off <<= 1) {
      s1[i] += __shfl_xor(s1[i], off);
      s2[i] += __shfl_xor(s2[i], off);
    }
    const float mean = s1[i] * (1.0f / 256.0f);
    const float var  = s2[i] * (1.0f / 256.0f) - mean * mean;
    s1[i] = mean;
    s2[i] = rsqrtf(var + 1e-5f);
  }
#pragma unroll
  for (int j = 0; j < 16; ++j) {
    const int n = j * 16 + r16;
    const float gg = gl[n], bb = bl[n];
#pragma unroll
    for (int i = 0; i < 8; ++i) {
      const int m = mb + hi * 8 + i;
      const float v = (acc[j][i] - s1[i]) * s2[i] * gg + bb;
      cat[(size_t)(token0 + m) * 512 + 256 + n] = (__bf16)v;
    }
  }
}

// ---------------------------------------------------------------------------
// msg2 = LN2( hid @ W2.T )  (fp32 out, [token, 256])
// ---------------------------------------------------------------------------
__global__ __launch_bounds__(256)
void mlp2_ln_kernel(const __bf16* __restrict__ A, const __bf16* __restrict__ Bw,
                    const float* __restrict__ g2, const float* __restrict__ b2,
                    float* __restrict__ msg2) {
  __shared__ __bf16 Alds[128 * 40];
  __shared__ __bf16 Blds[256 * 40];
  __shared__ float  gl[256], bl[256];

  const int tid  = threadIdx.x;
  const int wave = tid >> 5;
  const int lane = tid & 31;
  const int r16  = lane & 15;
  const int hi   = lane >> 4;
  const int token0 = blockIdx.x * 128;
  const int mb     = wave * 16;

  gl[tid] = g2[tid];
  bl[tid] = b2[tid];

  const v8f vz = {0.f, 0.f, 0.f, 0.f, 0.f, 0.f, 0.f, 0.f};
  v8f acc[16];
#pragma unroll
  for (int j = 0; j < 16; ++j) acc[j] = vz;

  for (int k0 = 0; k0 < 512; k0 += 32) {
    __syncthreads();
#pragma unroll
    for (int it = 0; it < 2; ++it) {
      int idx = tid * 2 + it;
      int r = idx >> 2, part = idx & 3;
      async_copy_b128(&Alds[r * 40 + part * 8],
                      A + (size_t)(token0 + r) * 512 + k0 + part * 8);
    }
#pragma unroll
    for (int it = 0; it < 4; ++it) {
      int idx = tid * 4 + it;
      int r = idx >> 2, part = idx & 3;
      async_copy_b128(&Blds[r * 40 + part * 8],
                      Bw + (size_t)r * 512 + k0 + part * 8);
    }
    async_wait0();
    __syncthreads();
    v16bf fa = frag_ld(&Alds[mb * 40], 40);
#pragma unroll
    for (int j = 0; j < 16; ++j) {
      v16bf fb = frag_ld(&Blds[j * 16 * 40], 40);
      acc[j] = WMMA_BF16(fa, fb, acc[j]);
    }
  }

  float s1[8] = {0, 0, 0, 0, 0, 0, 0, 0};
  float s2[8] = {0, 0, 0, 0, 0, 0, 0, 0};
#pragma unroll
  for (int j = 0; j < 16; ++j)
#pragma unroll
    for (int i = 0; i < 8; ++i) {
      const float v = acc[j][i];
      s1[i] += v;
      s2[i] += v * v;
    }
#pragma unroll
  for (int i = 0; i < 8; ++i) {
#pragma unroll
    for (int off = 1; off < 16; off <<= 1) {
      s1[i] += __shfl_xor(s1[i], off);
      s2[i] += __shfl_xor(s2[i], off);
    }
    const float mean = s1[i] * (1.0f / 256.0f);
    const float var  = s2[i] * (1.0f / 256.0f) - mean * mean;
    s1[i] = mean;
    s2[i] = rsqrtf(var + 1e-5f);
  }
#pragma unroll
  for (int j = 0; j < 16; ++j) {
    const int n = j * 16 + r16;
    const float gg = gl[n], bb = bl[n];
#pragma unroll
    for (int i = 0; i < 8; ++i) {
      const int m = mb + hi * 8 + i;
      msg2[(size_t)(token0 + m) * 256 + n] = (acc[j][i] - s1[i]) * s2[i] * gg + bb;
    }
  }
}

// ---------------------------------------------------------------------------
// Final: aug = BN(xt @ Ws.T + bs);  out[b,c,l] = msg2 + x + aug
// Stages 64-channel slabs through LDS so the [B,C,L] writes are coalesced.
// ---------------------------------------------------------------------------
__global__ __launch_bounds__(256)
void final_kernel(const __bf16* __restrict__ cat, const __bf16* __restrict__ Wsb,
                  const float* __restrict__ bs_c, const float* __restrict__ bn_g,
                  const float* __restrict__ bn_b, const float* __restrict__ bn_rm,
                  const float* __restrict__ bn_rv, const float* __restrict__ x,
                  const float* __restrict__ msg2, float* __restrict__ out) {
  __shared__ char smem[64 * 132 * 4];           // overlays GEMM tiles & stage
  __shared__ float scl[256], shf[256];
  __bf16* Alds = (__bf16*)smem;                  // 128*40 bf16 = 10240 B
  __bf16* Blds = (__bf16*)(smem + 128 * 40 * 2); // 256*40 bf16 = 20480 B
  float*  stage = (float*)smem;                  // 64*132 f32 (after K loop)

  const int tid  = threadIdx.x;
  const int wave = tid >> 5;
  const int lane = tid & 31;
  const int r16  = lane & 15;
  const int hi   = lane >> 4;
  const int token0 = blockIdx.x * 128;
  const int b      = token0 >> 13;
  const int l0     = token0 & (LL - 1);
  const int mb     = wave * 16;

  {
    const float rstd = rsqrtf(bn_rv[tid] + 1e-5f);
    const float a = rstd * bn_g[tid];
    scl[tid] = a;
    shf[tid] = (bs_c[tid] - bn_rm[tid]) * a + bn_b[tid];
  }

  const v8f vz = {0.f, 0.f, 0.f, 0.f, 0.f, 0.f, 0.f, 0.f};
  v8f acc[16];
#pragma unroll
  for (int j = 0; j < 16; ++j) acc[j] = vz;

  for (int k0 = 0; k0 < 256; k0 += 32) {
    __syncthreads();
#pragma unroll
    for (int it = 0; it < 2; ++it) {
      int idx = tid * 2 + it;
      int r = idx >> 2, part = idx & 3;
      async_copy_b128(&Alds[r * 40 + part * 8],
                      cat + (size_t)(token0 + r) * 512 + k0 + part * 8);
    }
#pragma unroll
    for (int it = 0; it < 4; ++it) {
      int idx = tid * 4 + it;
      int r = idx >> 2, part = idx & 3;
      async_copy_b128(&Blds[r * 40 + part * 8],
                      Wsb + (size_t)r * 256 + k0 + part * 8);
    }
    async_wait0();
    __syncthreads();
    v16bf fa = frag_ld(&Alds[mb * 40], 40);
#pragma unroll
    for (int j = 0; j < 16; ++j) {
      v16bf fb = frag_ld(&Blds[j * 16 * 40], 40);
      acc[j] = WMMA_BF16(fa, fb, acc[j]);
    }
  }
  __syncthreads();   // done with tiles; smem becomes the transpose stage

  for (int p = 0; p < 4; ++p) {                 // 64 channels per pass
#pragma unroll
    for (int jj = 0; jj < 4; ++jj) {
      const int j = p * 4 + jj;
      const int n = j * 16 + r16;               // global channel
      const int nl = n - p * 64;                // 0..63
      const float a = scl[n], c = shf[n];
#pragma unroll
      for (int i = 0; i < 8; ++i) {
        const int m = mb + hi * 8 + i;
        stage[nl * 132 + m] = acc[j][i] * a + c;   // BN(aug)
      }
    }
    __syncthreads();
    const int rid = tid >> 2, quad = tid & 3;
    const int o = p * 64 + rid;
    const float* xrow = x + ((size_t)b * CC + o) * LL + l0;
    float* orow = out + ((size_t)b * CC + o) * LL + l0;
#pragma unroll 4
    for (int mm = 0; mm < 32; ++mm) {
      const int m = quad * 32 + mm;
      orow[m] = stage[rid * 132 + m] + xrow[m] + msg2[(size_t)(token0 + m) * 256 + o];
    }
    __syncthreads();
  }
}

// ---------------------------------------------------------------------------
// Host orchestration
// ---------------------------------------------------------------------------
extern "C" void kernel_launch(void* const* d_in, const int* in_sizes, int n_in,
                              void* d_out, int out_size, void* d_ws, size_t ws_size,
                              hipStream_t stream) {
  (void)in_sizes; (void)n_in; (void)out_size; (void)ws_size;
  const float* x     = (const float*)d_in[0];
  const float* Wq    = (const float*)d_in[1];
  const float* bq    = (const float*)d_in[2];
  const float* Wk    = (const float*)d_in[3];
  const float* bk    = (const float*)d_in[4];
  const float* Wv    = (const float*)d_in[5];
  const float* bv    = (const float*)d_in[6];
  const float* thr   = (const float*)d_in[7];
  const float* W1    = (const float*)d_in[8];
  const float* W2    = (const float*)d_in[9];
  const float* ln1_g = (const float*)d_in[10];
  const float* ln1_b = (const float*)d_in[11];
  const float* ln2_g = (const float*)d_in[12];
  const float* ln2_b = (const float*)d_in[13];
  const float* Ws    = (const float*)d_in[14];
  const float* bs_c  = (const float*)d_in[15];
  const float* bn_g  = (const float*)d_in[16];
  const float* bn_b  = (const float*)d_in[17];
  const float* bn_rm = (const float*)d_in[18];
  const float* bn_rv = (const float*)d_in[19];
  float* out = (float*)d_out;

  char*  ws  = (char*)d_ws;
  size_t off = 0;
  auto alloc = [&](size_t bytes) -> void* {
    off = (off + 255) & ~(size_t)255;
    void* p = ws + off;
    off += bytes;
    return p;
  };

  __bf16* cat   = (__bf16*)alloc((size_t)MTOT * 512 * 2);  // [xt | ln1(msg)]
  __bf16* hid   = (__bf16*)alloc((size_t)MTOT * 512 * 2);
  __bf16* Qb    = (__bf16*)alloc((size_t)MTOT * 256 * 2);
  __bf16* Kb    = (__bf16*)alloc((size_t)MTOT * 256 * 2);
  __bf16* Vb    = (__bf16*)alloc((size_t)MTOT * 256 * 2);
  float*  msg2  = (float*)alloc((size_t)MTOT * 256 * 4);
  float*  KVbuf = (float*)alloc((size_t)(BB * HH * DD * DD + BB * 256) * 4);
  float*  Ksum  = KVbuf + BB * HH * DD * DD;               // contiguous for zeroing
  float*  Zbuf  = (float*)alloc((size_t)MTOT * HH * 4);
  __bf16* wqb   = (__bf16*)alloc(65536 * 2);
  __bf16* wkb   = (__bf16*)alloc(65536 * 2);
  __bf16* wvb   = (__bf16*)alloc(65536 * 2);
  __bf16* wsb   = (__bf16*)alloc(65536 * 2);
  __bf16* w1b   = (__bf16*)alloc(262144 * 2);
  __bf16* w2b   = (__bf16*)alloc(131072 * 2);

  // weights -> bf16
  cvt_f32_bf16_kernel<<<256, 256, 0, stream>>>(Wq, wqb, 65536);
  cvt_f32_bf16_kernel<<<256, 256, 0, stream>>>(Wk, wkb, 65536);
  cvt_f32_bf16_kernel<<<256, 256, 0, stream>>>(Wv, wvb, 65536);
  cvt_f32_bf16_kernel<<<256, 256, 0, stream>>>(Ws, wsb, 65536);
  cvt_f32_bf16_kernel<<<1024, 256, 0, stream>>>(W1, w1b, 262144);
  cvt_f32_bf16_kernel<<<512, 256, 0, stream>>>(W2, w2b, 131072);

  // zero KV + Ksum accumulators
  const int nz = BB * HH * DD * DD + BB * 256;
  zero_f32_kernel<<<(nz + 255) / 256, 256, 0, stream>>>(KVbuf, nz);

  // x [B,C,L] -> cat[:, 0:256] bf16 (token-major)
  transpose_kernel<<<dim3(LL / 32, CC / 32, BB), dim3(32, 8), 0, stream>>>(x, cat);

  // Q/K/V projections (fused ELU+1 / scaling; K also reduces Ksum)
  gemm_bf16_kernel<MODE_Q><<<dim3(MTOT / 128, 1), 256, 0, stream>>>(
      cat, 512, 256, wqb, bq, Qb, 256, nullptr);
  gemm_bf16_kernel<MODE_K><<<dim3(MTOT / 128, 1), 256, 0, stream>>>(
      cat, 512, 256, wkb, bk, Kb, 256, Ksum);
  gemm_bf16_kernel<MODE_V><<<dim3(MTOT / 128, 1), 256, 0, stream>>>(
      cat, 512, 256, wvb, bv, Vb, 256, nullptr);

  // KV = K^T V per (b,h)
  kv_kernel<<<dim3(BB * HH, LL / 1024), 256, 0, stream>>>(Kb, Vb, KVbuf);

  // Z = L / (sparsify(Q . Ksum) + eps)
  zscore_kernel<<<(MTOT * HH) / 256, 256, 0, stream>>>(Qb, Ksum, thr, Zbuf);

  // attention out + LN1 -> cat[:, 256:512]
  attn_kernel<<<MTOT / 128, 256, 0, stream>>>(Qb, KVbuf, Zbuf, ln1_g, ln1_b, cat);

  // MLP layer 1: hid = relu(cat @ W1.T)
  gemm_bf16_kernel<MODE_RELU><<<dim3(MTOT / 128, 2), 256, 0, stream>>>(
      cat, 512, 512, w1b, nullptr, hid, 512, nullptr);

  // MLP layer 2 + LN2 -> msg2 (fp32)
  mlp2_ln_kernel<<<MTOT / 128, 256, 0, stream>>>(hid, w2b, ln2_g, ln2_b, msg2);

  // shortcut GEMM + BN + residual sum -> out [B,C,L]
  final_kernel<<<MTOT / 128, 256, 0, stream>>>(cat, wsb, bs_c, bn_g, bn_b, bn_rm,
                                               bn_rv, x, msg2, out);
}
